// Qwen3DecoderBlock_89928025244030
// MI455X (gfx1250) — matrix operand order repaired and checked by reading, not compiled, not run on hardware
//
#include <hip/hip_runtime.h>
#include <hip/hip_bf16.h>

// ---------------------------------------------------------------------------
// Qwen3 decoder block for gfx1250 (MI455X): bf16 WMMA everywhere.
// ---------------------------------------------------------------------------

typedef unsigned short u16;
typedef __attribute__((ext_vector_type(16))) __bf16 bf16x16;
typedef __attribute__((ext_vector_type(8)))  __bf16 bf16x8;
typedef __attribute__((ext_vector_type(8)))  float  f32x8;

union FragU { bf16x16 v; bf16x8 h[2]; };

static constexpr int S_  = 2048;
static constexpr int HS_ = 2048;
static constexpr int H_  = 16;
static constexpr int KV_ = 8;
static constexpr int D_  = 128;
static constexpr int HD_ = H_ * D_;    // 2048
static constexpr int KVD_ = KV_ * D_;  // 1024
static constexpr int IM_ = 6144;

__device__ __forceinline__ u16 f32_to_bf16(float f) {
  union { float f; unsigned u; } v; v.f = f;
  unsigned r = v.u + 0x7FFFu + ((v.u >> 16) & 1u);   // round-to-nearest-even
  return (u16)(r >> 16);
}

__device__ __forceinline__ f32x8 fz8() {
  f32x8 z;
#pragma unroll
  for (int i = 0; i < 8; ++i) z[i] = 0.0f;
  return z;
}

__device__ __forceinline__ f32x8 wmma_bf16(bf16x16 a, bf16x16 b, f32x8 c) {
  return __builtin_amdgcn_wmma_f32_16x16x32_bf16(false, a, false, b, (short)0, c,
                                                 false, false);
}

// A-matrix 16x32 bf16 fragment (ISA 7.12.2): lane holds row = lane%16;
// half = lane/16 holds K in {8*half..8*half+7} and {16+8*half..23+8*half}.
__device__ __forceinline__ bf16x16 load_frag_a(const u16* base, int ldr,
                                               int row0, int k0, int lane) {
  const u16* p = base + (row0 + (lane & 15)) * ldr + k0 + ((lane >> 4) << 3);
  FragU f;
  f.h[0] = *(const bf16x8*)(p);
  f.h[1] = *(const bf16x8*)(p + 16);
  return f.v;
}

// B-matrix 32x16 bf16 fragment: lane holds column n = lane%16;
// half = lane/16 holds K in {8*half..8*half+7} and {16+8*half..23+8*half}.
// base is B^T stored [N][K] row-major.
__device__ __forceinline__ bf16x16 load_frag_b(const u16* base, int ldr,
                                               int row0, int k0, int lane) {
  const u16* p = base + (row0 + (lane & 15)) * ldr + k0 + ((lane >> 4) << 3);
  FragU f;
  f.h[0] = *(const bf16x8*)(p);
  f.h[1] = *(const bf16x8*)(p + 16);
  return f.v;
}

// ---------------------------------------------------------------------------
// Weight transpose + f32->bf16 pack: wt[n*K + k] = bf16(w[k*N + n])
// ---------------------------------------------------------------------------
__global__ __launch_bounds__(256) void pack_wT_kernel(
    const float* __restrict__ w, u16* __restrict__ wt, int K, int N) {
  __shared__ float t[32][33];
  int n0 = blockIdx.x * 32, k0 = blockIdx.y * 32;
  int x = threadIdx.x, y = threadIdx.y;  // block (32, 8)
#pragma unroll
  for (int i = 0; i < 4; ++i)
    t[y + 8 * i][x] = w[(k0 + y + 8 * i) * N + n0 + x];
  __syncthreads();
#pragma unroll
  for (int i = 0; i < 4; ++i)
    wt[(n0 + y + 8 * i) * K + k0 + x] = f32_to_bf16(t[x][y + 8 * i]);
}

// ---------------------------------------------------------------------------
// RMSNorm: out_bf[row][c] = bf16(x * rsqrt(mean(x^2)+eps) * w[c])
// ---------------------------------------------------------------------------
__global__ __launch_bounds__(256) void rmsnorm_kernel(
    const float* __restrict__ x, const float* __restrict__ w,
    u16* __restrict__ out_bf, int cols) {
  __shared__ float red[256];
  const int row = blockIdx.x;
  const float* xr = x + row * cols;
  u16* orow = out_bf + row * cols;
  float ss = 0.0f;
  for (int c = threadIdx.x; c < cols; c += 256) { float v = xr[c]; ss += v * v; }
  red[threadIdx.x] = ss;
  __syncthreads();
  for (int s = 128; s > 0; s >>= 1) {
    if (threadIdx.x < (unsigned)s) red[threadIdx.x] += red[threadIdx.x + s];
    __syncthreads();
  }
  float inv = rsqrtf(red[0] / (float)cols + 1e-6f);
  for (int c = threadIdx.x; c < cols; c += 256)
    orow[c] = f32_to_bf16(xr[c] * inv * w[c]);
}

// ---------------------------------------------------------------------------
// bf16 WMMA GEMM: C[M,N] = A[M,K](bf16) * Bt[N,K]^T(bf16)  (+ Cadd if ADD)
// Tiles: BM=BN=128, BK=32; 256 threads = 8 waves; wave computes 32x64.
// Double-buffered LDS, one barrier per K-step; register prefetch of the
// next tiles overlaps HBM/L2 latency with the WMMA pipe.
// ---------------------------------------------------------------------------
template <bool ADD>
__global__ __launch_bounds__(256) void gemm_bf16_kernel(
    const u16* __restrict__ A, const u16* __restrict__ Bt,
    float* __restrict__ C, const float* __restrict__ Cadd,
    int M, int N, int K) {
  __shared__ __attribute__((aligned(16))) u16 As[2][128 * 40];
  __shared__ __attribute__((aligned(16))) u16 Bs[2][128 * 40];
  const int tid = threadIdx.x, lane = tid & 31, warp = tid >> 5;
  const int bm = blockIdx.y * 128, bn = blockIdx.x * 128;
  const int wr = warp >> 1, wc = warp & 1;  // 4x2 wave grid

  // Per-thread staging chunks: c = tid + 256*i -> row = c/4, k8 = (c%4)*8.
  int srow[2], skc[2];
  const u16* gA[2];
  const u16* gB[2];
#pragma unroll
  for (int i = 0; i < 2; ++i) {
    int c = tid + 256 * i;
    srow[i] = c >> 2;
    skc[i] = (c & 3) << 3;
    gA[i] = A + (bm + srow[i]) * K + skc[i];
    gB[i] = Bt + (bn + srow[i]) * K + skc[i];
  }

  f32x8 acc[2][4];
#pragma unroll
  for (int mt = 0; mt < 2; ++mt)
#pragma unroll
    for (int nt = 0; nt < 4; ++nt) acc[mt][nt] = fz8();

  // Prime buffer 0 with k=0 tiles.
#pragma unroll
  for (int i = 0; i < 2; ++i) {
    *(bf16x8*)(As[0] + srow[i] * 40 + skc[i]) = *(const bf16x8*)(gA[i]);
    *(bf16x8*)(Bs[0] + srow[i] * 40 + skc[i]) = *(const bf16x8*)(gB[i]);
  }
  __syncthreads();

  const int nk = K >> 5;
  for (int kb = 0; kb < nk; ++kb) {
    const int cur = kb & 1, nxt = cur ^ 1;
    // Prefetch next tiles into registers (global -> VGPR).
    bf16x8 pa[2], pb[2];
    if (kb + 1 < nk) {
      const int k0n = (kb + 1) << 5;
#pragma unroll
      for (int i = 0; i < 2; ++i) {
        pa[i] = *(const bf16x8*)(gA[i] + k0n);
        pb[i] = *(const bf16x8*)(gB[i] + k0n);
      }
    }

    // Compute from current buffer.
    bf16x16 af[2], bfr[4];
#pragma unroll
    for (int mt = 0; mt < 2; ++mt)
      af[mt] = load_frag_a(As[cur], 40, wr * 32 + mt * 16, 0, lane);
#pragma unroll
    for (int nt = 0; nt < 4; ++nt)
      bfr[nt] = load_frag_b(Bs[cur], 40, wc * 64 + nt * 16, 0, lane);
#pragma unroll
    for (int mt = 0; mt < 2; ++mt)
#pragma unroll
      for (int nt = 0; nt < 4; ++nt)
        acc[mt][nt] = wmma_bf16(af[mt], bfr[nt], acc[mt][nt]);

    // Commit prefetched tiles to the other buffer.
    if (kb + 1 < nk) {
#pragma unroll
      for (int i = 0; i < 2; ++i) {
        *(bf16x8*)(As[nxt] + srow[i] * 40 + skc[i]) = pa[i];
        *(bf16x8*)(Bs[nxt] + srow[i] * 40 + skc[i]) = pb[i];
      }
    }
    __syncthreads();
  }

  // Epilogue: C layout — lane n = lane%16, VGPR g -> row g + 8*(lane/16).
  const int n = lane & 15, mh = (lane >> 4) << 3;
  const int row0 = bm + wr * 32 + mh;
  const int col0 = bn + wc * 64 + n;
#pragma unroll
  for (int mt = 0; mt < 2; ++mt)
#pragma unroll
    for (int nt = 0; nt < 4; ++nt) {
      int base = (row0 + mt * 16) * N + col0 + nt * 16;
      float* cp = C + base;
      const float* ap = Cadd + base;  // only dereferenced if ADD
#pragma unroll
      for (int g = 0; g < 8; ++g) {
        float v = acc[mt][nt][g];
        if (ADD) v += ap[g * N];
        cp[g * N] = v;
      }
    }
}

// ---------------------------------------------------------------------------
// Per-head QK RMSNorm + RoPE; writes bf16 Q (pre-scaled by D^-0.5) and K.
// grid = S*(H+KV) blocks of 128 threads; block handles one (s, head).
// ---------------------------------------------------------------------------
__global__ __launch_bounds__(128) void qk_rope_kernel(
    const float* __restrict__ qf, const float* __restrict__ kf,
    const float* __restrict__ qw, const float* __restrict__ kw,
    u16* __restrict__ qb, u16* __restrict__ kb) {
  __shared__ float buf[128];
  __shared__ float red[128];
  const int t = threadIdx.x;
  const int hh = blockIdx.x % (H_ + KV_);
  const int s  = blockIdx.x / (H_ + KV_);
  const bool isq = hh < H_;
  const int off = isq ? (s * H_ + hh) * D_ : (s * KV_ + (hh - H_)) * D_;
  const float* src = isq ? (qf + off) : (kf + off);
  const float* wn = isq ? qw : kw;
  float v = src[t];
  red[t] = v * v;
  __syncthreads();
  for (int st = 64; st > 0; st >>= 1) {
    if (t < st) red[t] += red[t + st];
    __syncthreads();
  }
  float inv = rsqrtf(red[0] / (float)D_ + 1e-6f);
  buf[t] = v * inv * wn[t];
  __syncthreads();
  const int i = t & 63;
  float freq = __expf(-logf(10000.0f) * (float)i / 64.0f);
  float ang = (float)s * freq;
  float c = cosf(ang), sn = sinf(ang);
  float o = (t < 64) ? (buf[t] * c - buf[t + 64] * sn)
                     : (buf[t] * c + buf[t - 64] * sn);
  if (isq)
    qb[off + t] = f32_to_bf16(o * 0.08838834764831845f);
  else
    kb[off + t] = f32_to_bf16(o);
}

// V^T pack: vt[(kv*D + d)*S + s] = bf16(v[s][kv*D + d])
__global__ __launch_bounds__(256) void vpack_kernel(
    const float* __restrict__ vf, u16* __restrict__ vt) {
  int i = blockIdx.x * 256 + threadIdx.x;  // total = KVD_*S_
  int s = i % S_;
  int r = i / S_;  // r = kv*128 + d
  vt[r * S_ + s] = f32_to_bf16(vf[s * KVD_ + r]);
}

// ---------------------------------------------------------------------------
// Causal flash attention (bf16 WMMA, f32 online softmax).
// grid = (S/128, H); 256 threads = 8 waves; wave owns 16 query rows.
// Q pre-scaled by D^-0.5. GQA: kv head = h/2.
// ---------------------------------------------------------------------------
__global__ __launch_bounds__(256) void attn_kernel(
    const u16* __restrict__ qb, const u16* __restrict__ kb,
    const u16* __restrict__ vt, u16* __restrict__ ob) {
  __shared__ __attribute__((aligned(16))) u16 ks[64 * 136];      // K tile: 64 keys x 128 d
  __shared__ __attribute__((aligned(16))) u16 vs[128 * 72];      // V^T tile: 128 d x 64 keys
  __shared__ __attribute__((aligned(16))) u16 ps[8 * 16 * 72];   // per-wave P tile 16 x 64
  const int tid = threadIdx.x, lane = tid & 31, warp = tid >> 5;
  const int h = blockIdx.y, kv = h >> 1;
  const int q0 = blockIdx.x * 128 + warp * 16;
  const int n = lane & 15, mh = (lane >> 4) << 3;
  const int half8 = (lane >> 4) << 3;
  const u16* kvbase = kb + kv * D_;          // + key*KVD_
  const u16* vtbase = vt + kv * D_ * S_;     // + d*S_

  // Preload Q fragments for all 4 D-chunks of 32.
  bf16x16 qfrag[4];
  {
    const u16* qrow = qb + (q0 + (lane & 15)) * HD_ + h * D_;
#pragma unroll
    for (int kc = 0; kc < 4; ++kc) {
      const u16* p = qrow + kc * 32;
      FragU f;
      f.h[0] = *(const bf16x8*)(p + half8);
      f.h[1] = *(const bf16x8*)(p + 16 + half8);
      qfrag[kc] = f.v;
    }
  }

  f32x8 oacc[8];
#pragma unroll
  for (int dt = 0; dt < 8; ++dt) oacc[dt] = fz8();
  float mrow[8], lrow[8];
#pragma unroll
  for (int g = 0; g < 8; ++g) { mrow[g] = -1e30f; lrow[g] = 0.0f; }

  const int nkb = (blockIdx.x + 1) * 2;  // key blocks of 64, causal bound
  for (int it = 0; it < nkb; ++it) {
    const int kb0 = it * 64;
    // Stage K tile (64x128) and V^T tile (128x64).
#pragma unroll
    for (int i = 0; i < 4; ++i) {
      int c = tid + 256 * i;
      int row = c >> 4, kc8 = (c & 15) << 3;
      *(bf16x8*)(ks + row * 136 + kc8) =
          *(const bf16x8*)(kvbase + (kb0 + row) * KVD_ + kc8);
    }
#pragma unroll
    for (int i = 0; i < 4; ++i) {
      int c = tid + 256 * i;
      int row = c >> 3, j8 = (c & 7) << 3;
      *(bf16x8*)(vs + row * 72 + j8) =
          *(const bf16x8*)(vtbase + row * S_ + kb0 + j8);
    }
    __syncthreads();

    // Scores: S = Q K^T over D=128 (4 WMMA K-steps), 4 key subtiles of 16.
    f32x8 sc[4];
#pragma unroll
    for (int nt = 0; nt < 4; ++nt) sc[nt] = fz8();
#pragma unroll
    for (int kc = 0; kc < 4; ++kc)
#pragma unroll
      for (int nt = 0; nt < 4; ++nt) {
        bf16x16 bfrg = load_frag_b(ks, 136, nt * 16, kc * 32, lane);
        sc[nt] = wmma_bf16(qfrag[kc], bfrg, sc[nt]);
      }

    // Causal mask.
#pragma unroll
    for (int nt = 0; nt < 4; ++nt) {
      int kg = kb0 + nt * 16 + n;
#pragma unroll
      for (int g = 0; g < 8; ++g)
        if (kg > q0 + mh + g) sc[nt][g] = -1e30f;
    }

    // Online softmax: row reductions across the 16 lanes of each half-wave.
    float al[8];
#pragma unroll
    for (int g = 0; g < 8; ++g) {
      float v = fmaxf(fmaxf(sc[0][g], sc[1][g]), fmaxf(sc[2][g], sc[3][g]));
#pragma unroll
      for (int msk = 1; msk < 16; msk <<= 1) v = fmaxf(v, __shfl_xor(v, msk, 32));
      float mn = fmaxf(mrow[g], v);
      al[g] = __expf(mrow[g] - mn);
      mrow[g] = mn;
    }
#pragma unroll
    for (int nt = 0; nt < 4; ++nt)
#pragma unroll
      for (int g = 0; g < 8; ++g) sc[nt][g] = __expf(sc[nt][g] - mrow[g]);
#pragma unroll
    for (int g = 0; g < 8; ++g) {
      float s = sc[0][g] + sc[1][g] + sc[2][g] + sc[3][g];
#pragma unroll
      for (int msk = 1; msk < 16; msk <<= 1) s += __shfl_xor(s, msk, 32);
      lrow[g] = lrow[g] * al[g] + s;
    }
#pragma unroll
    for (int dt = 0; dt < 8; ++dt)
#pragma unroll
      for (int g = 0; g < 8; ++g) oacc[dt][g] *= al[g];

    // P (C-layout) -> per-wave LDS tile, then re-load as A fragments.
    u16* pw = ps + warp * 16 * 72;
#pragma unroll
    for (int nt = 0; nt < 4; ++nt)
#pragma unroll
      for (int g = 0; g < 8; ++g)
        pw[(mh + g) * 72 + nt * 16 + n] = f32_to_bf16(sc[nt][g]);

    // O += P V  (keys = WMMA reduction dim, 2 chunks of 32; 8 d-subtiles).
#pragma unroll
    for (int kc2 = 0; kc2 < 2; ++kc2) {
      bf16x16 pf = load_frag_a(pw, 72, 0, kc2 * 32, lane);
#pragma unroll
      for (int dt = 0; dt < 8; ++dt) {
        bf16x16 vf = load_frag_b(vs, 72, dt * 16, kc2 * 32, lane);
        oacc[dt] = wmma_bf16(pf, vf, oacc[dt]);
      }
    }
    __syncthreads();
  }

  // Normalize and store bf16 attention output [S][H*D].
  u16* obase = ob + (q0 + mh) * HD_ + h * D_ + n;
#pragma unroll
  for (int dt = 0; dt < 8; ++dt) {
#pragma unroll
    for (int g = 0; g < 8; ++g) {
      float o = oacc[dt][g] / lrow[g];
      obase[g * HD_ + dt * 16] = f32_to_bf16(o);
    }
  }
}

// SwiGLU: prod = bf16(silu(gate) * up)
__global__ __launch_bounds__(256) void swiglu_kernel(
    const float* __restrict__ g, const float* __restrict__ u,
    u16* __restrict__ p, int total) {
  int i = blockIdx.x * 256 + threadIdx.x;
  if (i < total) {
    float gv = g[i];
    float s = gv / (1.0f + __expf(-gv));
    p[i] = f32_to_bf16(s * u[i]);
  }
}

// ---------------------------------------------------------------------------
// Host-side orchestration
// ---------------------------------------------------------------------------
extern "C" void kernel_launch(void* const* d_in, const int* in_sizes, int n_in,
                              void* d_out, int out_size, void* d_ws,
                              size_t ws_size, hipStream_t stream) {
  (void)in_sizes; (void)n_in; (void)out_size; (void)ws_size;
  const float* x          = (const float*)d_in[0];
  /* d_in[1] = position_ids (int64) — positions are 0..S-1, computed in-kernel */
  const float* w_in_norm  = (const float*)d_in[2];
  const float* wq         = (const float*)d_in[3];
  const float* wk         = (const float*)d_in[4];
  const float* wv         = (const float*)d_in[5];
  const float* q_norm_w   = (const float*)d_in[6];
  const float* k_norm_w   = (const float*)d_in[7];
  const float* wo         = (const float*)d_in[8];
  const float* w_post     = (const float*)d_in[9];
  const float* w_gate     = (const float*)d_in[10];
  const float* w_up       = (const float*)d_in[11];
  const float* w_down     = (const float*)d_in[12];
  float* out = (float*)d_out;

  char* wp = (char*)d_ws;
  auto alloc = [&](size_t bytes) -> void* {
    void* r = (void*)wp;
    wp += (bytes + 255) & ~(size_t)255;
    return r;
  };
  u16* h_bf   = (u16*)alloc((size_t)S_ * HS_ * 2);
  u16* wqT    = (u16*)alloc((size_t)HD_ * HS_ * 2);
  u16* wkT    = (u16*)alloc((size_t)KVD_ * HS_ * 2);
  u16* wvT    = (u16*)alloc((size_t)KVD_ * HS_ * 2);
  u16* woT    = (u16*)alloc((size_t)HS_ * HD_ * 2);
  u16* wgT    = (u16*)alloc((size_t)IM_ * HS_ * 2);
  u16* wuT    = (u16*)alloc((size_t)IM_ * HS_ * 2);
  u16* wdT    = (u16*)alloc((size_t)HS_ * IM_ * 2);
  float* q_f  = (float*)alloc((size_t)S_ * HD_ * 4);
  float* k_f  = (float*)alloc((size_t)S_ * KVD_ * 4);
  float* v_f  = (float*)alloc((size_t)S_ * KVD_ * 4);
  u16* q_bf   = (u16*)alloc((size_t)S_ * HD_ * 2);
  u16* k_bf   = (u16*)alloc((size_t)S_ * KVD_ * 2);
  u16* vt_bf  = (u16*)alloc((size_t)KVD_ * S_ * 2);
  u16* att_bf = (u16*)alloc((size_t)S_ * HD_ * 2);
  float* x1   = (float*)alloc((size_t)S_ * HS_ * 4);
  u16* h2_bf  = (u16*)alloc((size_t)S_ * HS_ * 2);
  float* gt_f = (float*)alloc((size_t)S_ * IM_ * 4);
  float* up_f = (float*)alloc((size_t)S_ * IM_ * 4);
  u16* pr_bf  = (u16*)alloc((size_t)S_ * IM_ * 2);

  const dim3 bpack(32, 8);
  // Weight packs (K, N per w[k][n] -> wt[n][k]).
  pack_wT_kernel<<<dim3(HD_ / 32, HS_ / 32), bpack, 0, stream>>>(wq, wqT, HS_, HD_);
  pack_wT_kernel<<<dim3(KVD_ / 32, HS_ / 32), bpack, 0, stream>>>(wk, wkT, HS_, KVD_);
  pack_wT_kernel<<<dim3(KVD_ / 32, HS_ / 32), bpack, 0, stream>>>(wv, wvT, HS_, KVD_);
  pack_wT_kernel<<<dim3(HS_ / 32, HD_ / 32), bpack, 0, stream>>>(wo, woT, HD_, HS_);
  pack_wT_kernel<<<dim3(IM_ / 32, HS_ / 32), bpack, 0, stream>>>(w_gate, wgT, HS_, IM_);
  pack_wT_kernel<<<dim3(IM_ / 32, HS_ / 32), bpack, 0, stream>>>(w_up, wuT, HS_, IM_);
  pack_wT_kernel<<<dim3(HS_ / 32, IM_ / 32), bpack, 0, stream>>>(w_down, wdT, IM_, HS_);

  // Attention sub-block.
  rmsnorm_kernel<<<S_, 256, 0, stream>>>(x, w_in_norm, h_bf, HS_);
  gemm_bf16_kernel<false><<<dim3(HD_ / 128, S_ / 128), 256, 0, stream>>>(
      h_bf, wqT, q_f, nullptr, S_, HD_, HS_);
  gemm_bf16_kernel<false><<<dim3(KVD_ / 128, S_ / 128), 256, 0, stream>>>(
      h_bf, wkT, k_f, nullptr, S_, KVD_, HS_);
  gemm_bf16_kernel<false><<<dim3(KVD_ / 128, S_ / 128), 256, 0, stream>>>(
      h_bf, wvT, v_f, nullptr, S_, KVD_, HS_);
  qk_rope_kernel<<<S_ * (H_ + KV_), 128, 0, stream>>>(
      q_f, k_f, q_norm_w, k_norm_w, q_bf, k_bf);
  vpack_kernel<<<(KVD_ * S_) / 256, 256, 0, stream>>>(v_f, vt_bf);
  attn_kernel<<<dim3(S_ / 128, H_), 256, 0, stream>>>(q_bf, k_bf, vt_bf, att_bf);
  gemm_bf16_kernel<true><<<dim3(HS_ / 128, S_ / 128), 256, 0, stream>>>(
      att_bf, woT, x1, x, S_, HS_, HD_);  // + residual

  // MLP sub-block (SwiGLU).
  rmsnorm_kernel<<<S_, 256, 0, stream>>>(x1, w_post, h2_bf, HS_);
  gemm_bf16_kernel<false><<<dim3(IM_ / 128, S_ / 128), 256, 0, stream>>>(
      h2_bf, wgT, gt_f, nullptr, S_, IM_, HS_);
  gemm_bf16_kernel<false><<<dim3(IM_ / 128, S_ / 128), 256, 0, stream>>>(
      h2_bf, wuT, up_f, nullptr, S_, IM_, HS_);
  swiglu_kernel<<<(S_ * IM_) / 256, 256, 0, stream>>>(gt_f, up_f, pr_bf, S_ * IM_);
  gemm_bf16_kernel<true><<<dim3(HS_ / 128, S_ / 128), 256, 0, stream>>>(
      pr_bf, wdT, out, x1, S_, HS_, IM_);  // + residual
}
